// SpatialDependencyPerception_75453985457579
// MI455X (gfx1250) — matrix operand order repaired and checked by reading, not compiled, not run on hardware
//
#include <hip/hip_runtime.h>
#include <hip/hip_bf16.h>

typedef __attribute__((ext_vector_type(16))) __bf16 v16bf;
typedef __attribute__((ext_vector_type(8)))  __bf16 v8bf;
typedef __attribute__((ext_vector_type(8)))  float  v8f;

#define BATCH 8
#define CCH   64
#define HH    256
#define WW    256
#define HWSZ  65536   // H*W
#define NGRP  32
#define NWIN  32      // W / 8
#define EPSV  1e-5f
#define LSTR  72      // bf16 LDS row stride: 144B rows -> 16B aligned, 36-dword
                      // lane stride (9 coprime 16 -> conflict-free columns)
#define OSTR  68      // float LDS row stride: 272B rows -> 16B aligned,
                      // 4l mod 64 distinct bank groups across 16 lanes

// Assemble a WMMA A fragment (16x32, this code's k = 16h + s order, as B): this lane loads 16
// contiguous elements [kb..kb+15] of its row -> two aligned b128 LDS loads.
__device__ __forceinline__ v16bf lds_fragA(const __bf16* p) {
    v8bf lo = *(const v8bf*)(p);
    v8bf hi = *(const v8bf*)(p + 8);
    return __builtin_shufflevector(lo, hi, 0, 1, 2, 3, 4, 5, 6, 7,
                                           8, 9, 10, 11, 12, 13, 14, 15);
}
// Assemble a WMMA B fragment (32x16, this code's k = 16h + s order, as A): this lane loads 16
// contiguous elements -> two aligned b128 LDS loads.
__device__ __forceinline__ v16bf lds_fragB(const __bf16* p) {
    v8bf lo = *(const v8bf*)(p);
    v8bf hi = *(const v8bf*)(p + 8);
    return __builtin_shufflevector(lo, hi, 0, 1, 2, 3, 4, 5, 6, 7,
                                           8, 9, 10, 11, 12, 13, 14, 15);
}
// Pack two floats into a bf16x2 dword (v_cvt_pk_bf16_f32 friendly).
__device__ __forceinline__ unsigned pkbf(float a, float b) {
    union { __bf16 h[2]; unsigned u; } u_;
    u_.h[0] = (__bf16)a;
    u_.h[1] = (__bf16)b;
    return u_.u;
}

// ---------------------------------------------------------------------------
// Kernel 0: zero the stats accumulators (workspace is poisoned by harness)
// ---------------------------------------------------------------------------
__global__ void sdp_zero_kernel(float* __restrict__ p, int n) {
    int i = blockIdx.x * blockDim.x + threadIdx.x;
    if (i < n) p[i] = 0.0f;
}

// ---------------------------------------------------------------------------
// Kernel 1: 1x1 conv (bf16 WMMA) + GroupNorm partial stats.
//   out[b,o,hw] = sum_c W[o,c] * x[b,c,hw]   (M=o, K=c, N=hw)
//   grid = (HW/128, B, 2)  z: 0 -> (x_low,Wq,qhat)  1 -> (x_high,Wk,khat)
//   block = 256 (8 waves, 16 pixels each)
// ---------------------------------------------------------------------------
__global__ __launch_bounds__(256) void sdp_conv_stats_kernel(
    const float* __restrict__ x_low, const float* __restrict__ x_high,
    const float* __restrict__ Wq,    const float* __restrict__ Wk,
    __bf16* __restrict__ qhat, __bf16* __restrict__ khat,
    float* __restrict__ stats /* [2][B][G][2] */)
{
    const int z = blockIdx.z, b = blockIdx.y;
    const float* __restrict__ x  = (z ? x_high : x_low) + (size_t)b * CCH * HWSZ;
    const float* __restrict__ Wm = (z ? Wk : Wq);
    __bf16* __restrict__ outp    = (z ? khat : qhat) + (size_t)b * CCH * HWSZ;
    float* __restrict__ st       = stats + ((size_t)z * BATCH + b) * (NGRP * 2);

    __shared__ __align__(16) __bf16 Wlds[CCH * LSTR];
    __shared__ float sred[NGRP * 2];

    const int tid = threadIdx.x;
    for (int i = tid; i < CCH * CCH; i += 256) {
        const int o = i >> 6, c = i & 63;
        Wlds[o * LSTR + c] = (__bf16)Wm[i];
    }
    if (tid < NGRP * 2) sred[tid] = 0.0f;
    __syncthreads();

    const int wv     = tid >> 5;
    const int lane   = tid & 31;
    const int lane15 = lane & 15;
    const int half8  = (lane >> 4) * 8;   // C-row offset
    const int khalf  = (lane >> 4) * 16;  // A/B K base (this code's contiguous k = 16h + s)
    const int pixbase = blockIdx.x * 128 + wv * 16;

    v8f acc[4] = {};
#pragma unroll
    for (int kstep = 0; kstep < 2; ++kstep) {
        // B tile: B[k=c][n=pixel] = x[c][pixbase+n]
        v16bf Bm;
        const int cbase = kstep * 32 + khalf;
        const float* __restrict__ xp = x + pixbase + lane15;
#pragma unroll
        for (int v = 0; v < 8; ++v) {
            Bm[2 * v]     = (__bf16)xp[(size_t)(cbase + 2 * v) * HWSZ];
            Bm[2 * v + 1] = (__bf16)xp[(size_t)(cbase + 2 * v + 1) * HWSZ];
        }
        // streaming prefetch for the next 128-pixel block of this channel slab
        __builtin_prefetch(xp + (size_t)cbase * HWSZ + 128, 0, 0);
        __builtin_prefetch(xp + (size_t)(cbase + 16) * HWSZ + 128, 0, 0);
#pragma unroll
        for (int mt = 0; mt < 4; ++mt) {
            const int o  = mt * 16 + lane15;
            const int kb = kstep * 32 + khalf;
            v16bf Am = lds_fragA(&Wlds[o * LSTR + kb]);
            acc[mt] = __builtin_amdgcn_wmma_f32_16x16x32_bf16(
                false, Am, false, Bm, (short)0, acc[mt], false, false);
        }
    }

    // store bf16 result: C layout -> channel o = mt*16 + half8 + r, pixel = pixbase + lane15
#pragma unroll
    for (int mt = 0; mt < 4; ++mt) {
#pragma unroll
        for (int r = 0; r < 8; ++r) {
            const int o = mt * 16 + half8 + r;
            outp[(size_t)o * HWSZ + pixbase + lane15] = (__bf16)acc[mt][r];
        }
    }

    // GroupNorm partials: channel pair (2j,2j+1) within this lane's half -> group
#pragma unroll
    for (int mt = 0; mt < 4; ++mt) {
#pragma unroll
        for (int j = 0; j < 4; ++j) {
            const float a0 = acc[mt][2 * j], a1 = acc[mt][2 * j + 1];
            float s  = a0 + a1;
            float ss = a0 * a0 + a1 * a1;
#pragma unroll
            for (int m = 8; m >= 1; m >>= 1) {   // reduce across 16 lanes of this half
                s  += __shfl_xor(s,  m, 32);
                ss += __shfl_xor(ss, m, 32);
            }
            if (lane15 == 0) {
                const int g = mt * 8 + (half8 >> 1) + j;
                atomicAdd(&sred[g * 2 + 0], s);
                atomicAdd(&sred[g * 2 + 1], ss);
            }
        }
    }
    __syncthreads();
    if (tid < NGRP * 2) atomicAdd(&st[tid], sred[tid]);
}

// ---------------------------------------------------------------------------
// Kernel 2: finalize mean / rstd per (z,b,g)
// ---------------------------------------------------------------------------
__global__ void sdp_finalize_kernel(const float* __restrict__ stats,
                                    float* __restrict__ mr) {
    const int i = threadIdx.x;               // 0..511 = (z,b,g)
    if (i < 2 * BATCH * NGRP) {
        const float N    = 2.0f * (float)HWSZ;   // (C/G) * H * W
        const float s    = stats[i * 2 + 0];
        const float ss   = stats[i * 2 + 1];
        const float mean = s / N;
        const float var  = ss / N - mean * mean;
        mr[i * 2 + 0] = mean;
        mr[i * 2 + 1] = rsqrtf(var + EPSV);
    }
}

// ---------------------------------------------------------------------------
// Kernel 3: windowed cross-attention, bf16 WMMA, fused GN affine + residual.
//   grid = (1024 windows, B), block = 128 (4 waves; wave = one 16-row M tile)
//   Residual x_low window is fetched by the async-DMA path
//   (global_load_async_to_lds_b128, ASYNCcnt) issued up-front and drained
//   after both GEMMs.
// ---------------------------------------------------------------------------
__global__ __launch_bounds__(128) void sdp_attn_kernel(
    const float*  __restrict__ x_low,
    const __bf16* __restrict__ qhat, const __bf16* __restrict__ khat,
    const float* __restrict__ gq, const float* __restrict__ bq,
    const float* __restrict__ gk, const float* __restrict__ bk,
    const float* __restrict__ mr /* [2][B][G][2] */,
    float* __restrict__ outp)
{
    const int l = blockIdx.x, b = blockIdx.y;
    const int wy = l >> 5, wx = l & (NWIN - 1);

    __shared__ __align__(16) __bf16 Qs[64 * LSTR];  // [pixel][ch]
    __shared__ __align__(16) __bf16 Ks[64 * LSTR];  // [pixel][ch]
    __shared__ __align__(16) __bf16 Kt[CCH * LSTR]; // [ch][pixel] (for O = P*K)
    __shared__ __align__(16) __bf16 Ps[64 * LSTR];  // [qpixel][kpixel]
    __shared__ __align__(16) float Ost[CCH * OSTR]; // O transposed: [ch][pixel]
    __shared__ __align__(16) float Rs[CCH * 64];    // raw x_low window [ch][pixel]

    const int tid = threadIdx.x;
    const size_t base = (size_t)b * CCH * HWSZ + (size_t)wy * 8 * WW + (size_t)wx * 8;

    // ---- kick off async residual copy: global x_low window -> Rs ----------
    for (int idx = tid; idx < CCH * 8; idx += 128) {
        const int c = idx >> 3, py = idx & 7;
        const float* gp = x_low + base + (size_t)c * HWSZ + (size_t)py * WW;
        unsigned ldsoff = (unsigned)(uintptr_t)(&Rs[c * 64 + py * 8]);
        asm volatile("global_load_async_to_lds_b128 %0, %1, off"
                     :: "v"(ldsoff), "v"(gp) : "memory");
        asm volatile("global_load_async_to_lds_b128 %0, %1, off offset:16"
                     :: "v"(ldsoff), "v"(gp) : "memory");
    }

    const float* __restrict__ mrq = mr + (size_t)b * NGRP * 2;
    const float* __restrict__ mrk = mr + (size_t)(BATCH + b) * NGRP * 2;

    // ---- stage Q, K windows: GN affine on a channel PAIR (== one group),
    //      packed bf16x2 dword stores, 32 lanes -> consecutive LDS dwords ----
    for (int idx = tid; idx < NGRP * 8; idx += 128) {
        const int cp = idx & 31, py = idx >> 5;     // channel pair == group
        const int c0 = cp * 2, c1 = c0 + 1;
        const size_t off0 = base + (size_t)c0 * HWSZ + (size_t)py * WW;
        const size_t off1 = off0 + HWSZ;
        {
            const v8bf q0 = *(const v8bf*)(qhat + off0);
            const v8bf q1 = *(const v8bf*)(qhat + off1);
            const float mean = mrq[cp * 2], rstd = mrq[cp * 2 + 1];
            const float ga0 = gq[c0], be0 = bq[c0];
            const float ga1 = gq[c1], be1 = bq[c1];
#pragma unroll
            for (int px = 0; px < 8; ++px) {
                const float a0 = ((float)q0[px] - mean) * rstd * ga0 + be0;
                const float a1 = ((float)q1[px] - mean) * rstd * ga1 + be1;
                *(unsigned*)(&Qs[(py * 8 + px) * LSTR + c0]) = pkbf(a0, a1);
            }
        }
        {
            const v8bf k0 = *(const v8bf*)(khat + off0);
            const v8bf k1 = *(const v8bf*)(khat + off1);
            const float mean = mrk[cp * 2], rstd = mrk[cp * 2 + 1];
            const float ga0 = gk[c0], be0 = bk[c0];
            const float ga1 = gk[c1], be1 = bk[c1];
            v8bf kt0, kt1;
#pragma unroll
            for (int px = 0; px < 8; ++px) {
                const float a0 = ((float)k0[px] - mean) * rstd * ga0 + be0;
                const float a1 = ((float)k1[px] - mean) * rstd * ga1 + be1;
                const __bf16 h0 = (__bf16)a0, h1 = (__bf16)a1;
                kt0[px] = h0;
                kt1[px] = h1;
                union { __bf16 h[2]; unsigned u; } uu;
                uu.h[0] = h0; uu.h[1] = h1;
                *(unsigned*)(&Ks[(py * 8 + px) * LSTR + c0]) = uu.u;
            }
            *(v8bf*)(&Kt[c0 * LSTR + py * 8]) = kt0;   // transposed copies,
            *(v8bf*)(&Kt[c1 * LSTR + py * 8]) = kt1;   // 16B stores
        }
    }
    __syncthreads();

    const int wv     = tid >> 5;
    const int lane   = tid & 31;
    const int lane15 = lane & 15;
    const int half8  = (lane >> 4) * 8;
    const int khalf  = (lane >> 4) * 16;

    // ---- S = Q * K^T  (contract over channels) ----
    v8f accS[4] = {};
#pragma unroll
    for (int kstep = 0; kstep < 2; ++kstep) {
        const int m  = wv * 16 + lane15;
        const int kb = kstep * 32 + khalf;
        v16bf Am = lds_fragA(&Qs[m * LSTR + kb]);
#pragma unroll
        for (int nt = 0; nt < 4; ++nt) {
            const int kp = nt * 16 + lane15;           // B[c][keypix] = Ks[keypix][c]
            const int cb = kstep * 32 + khalf;
            v16bf Bm = lds_fragB(&Ks[kp * LSTR + cb]);
            accS[nt] = __builtin_amdgcn_wmma_f32_16x16x32_bf16(
                false, Am, false, Bm, (short)0, accS[nt], false, false);
        }
    }

    // ---- row softmax (registers + shfl within 16-lane halves) ----
#pragma unroll
    for (int r = 0; r < 8; ++r) {
        float v0 = accS[0][r] * 0.125f;
        float v1 = accS[1][r] * 0.125f;
        float v2 = accS[2][r] * 0.125f;
        float v3 = accS[3][r] * 0.125f;
        float mx = fmaxf(fmaxf(v0, v1), fmaxf(v2, v3));
#pragma unroll
        for (int m = 8; m >= 1; m >>= 1) mx = fmaxf(mx, __shfl_xor(mx, m, 32));
        const float e0 = __expf(v0 - mx), e1 = __expf(v1 - mx);
        const float e2 = __expf(v2 - mx), e3 = __expf(v3 - mx);
        float sum = (e0 + e1) + (e2 + e3);
#pragma unroll
        for (int m = 8; m >= 1; m >>= 1) sum += __shfl_xor(sum, m, 32);
        const float inv = 1.0f / sum;
        const int mrow = wv * 16 + half8 + r;
        Ps[mrow * LSTR +  0 + lane15] = (__bf16)(e0 * inv);
        Ps[mrow * LSTR + 16 + lane15] = (__bf16)(e1 * inv);
        Ps[mrow * LSTR + 32 + lane15] = (__bf16)(e2 * inv);
        Ps[mrow * LSTR + 48 + lane15] = (__bf16)(e3 * inv);
    }
    __syncthreads();

    // ---- O = P * K  (contract over key pixels; B from transposed Kt) ----
    v8f accO[4] = {};
#pragma unroll
    for (int kstep = 0; kstep < 2; ++kstep) {
        const int m  = wv * 16 + lane15;
        const int kb = kstep * 32 + khalf;
        v16bf Am = lds_fragA(&Ps[m * LSTR + kb]);
#pragma unroll
        for (int nt = 0; nt < 4; ++nt) {
            const int c   = nt * 16 + lane15;          // B[keypix][c] = Kt[c][keypix]
            const int kb2 = kstep * 32 + khalf;
            v16bf Bm = lds_fragB(&Kt[c * LSTR + kb2]);
            accO[nt] = __builtin_amdgcn_wmma_f32_16x16x32_bf16(
                false, Am, false, Bm, (short)0, accO[nt], false, false);
        }
    }

    // park O transposed: each lane owns 8 consecutive pixels of one channel
    // -> two float4 LDS stores per tile
#pragma unroll
    for (int nt = 0; nt < 4; ++nt) {
        const int c = nt * 16 + lane15;
        float* p = &Ost[c * OSTR + wv * 16 + half8];
        float4 lo, hi;
        lo.x = accO[nt][0]; lo.y = accO[nt][1]; lo.z = accO[nt][2]; lo.w = accO[nt][3];
        hi.x = accO[nt][4]; hi.y = accO[nt][5]; hi.z = accO[nt][6]; hi.w = accO[nt][7];
        ((float4*)p)[0] = lo;
        ((float4*)p)[1] = hi;
    }
    __syncthreads();

    // drain the async residual DMA (self-issued chunks, per-wave wait)
    asm volatile("s_wait_asynccnt 0x0" ::: "memory");

    // residual add + coalesced store in [B,C,H,W] layout (all float4)
    for (int idx = tid; idx < CCH * 8; idx += 128) {
        const int c = idx >> 3, py = idx & 7;
        const size_t off = base + (size_t)c * HWSZ + (size_t)py * WW;
        const float4 r0 = ((const float4*)(&Rs[c * 64 + py * 8]))[0];
        const float4 r1 = ((const float4*)(&Rs[c * 64 + py * 8]))[1];
        const float4 s0 = ((const float4*)(&Ost[c * OSTR + py * 8]))[0];
        const float4 s1 = ((const float4*)(&Ost[c * OSTR + py * 8]))[1];
        float4 o0, o1;
        o0.x = s0.x + r0.x; o0.y = s0.y + r0.y; o0.z = s0.z + r0.z; o0.w = s0.w + r0.w;
        o1.x = s1.x + r1.x; o1.y = s1.y + r1.y; o1.z = s1.z + r1.z; o1.w = s1.w + r1.w;
        ((float4*)(outp + off))[0] = o0;
        ((float4*)(outp + off))[1] = o1;
    }
}

// ---------------------------------------------------------------------------
extern "C" void kernel_launch(void* const* d_in, const int* in_sizes, int n_in,
                              void* d_out, int out_size, void* d_ws, size_t ws_size,
                              hipStream_t stream) {
    const float* x_low  = (const float*)d_in[0];
    const float* x_high = (const float*)d_in[1];
    const float* Wq     = (const float*)d_in[2];
    const float* Wk     = (const float*)d_in[3];
    const float* gq     = (const float*)d_in[4];
    const float* bq     = (const float*)d_in[5];
    const float* gk     = (const float*)d_in[6];
    const float* bk     = (const float*)d_in[7];
    float* out = (float*)d_out;

    char* ws = (char*)d_ws;
    const size_t nElem = (size_t)BATCH * CCH * HWSZ;     // 33,554,432
    __bf16* qhat = (__bf16*)ws;
    __bf16* khat = qhat + nElem;
    float*  stats = (float*)(ws + 2 * nElem * sizeof(__bf16));
    float*  mr    = stats + 2 * BATCH * NGRP * 2;

    sdp_zero_kernel<<<4, 256, 0, stream>>>(stats, 2 * BATCH * NGRP * 2);

    sdp_conv_stats_kernel<<<dim3(HWSZ / 128, BATCH, 2), 256, 0, stream>>>(
        x_low, x_high, Wq, Wk, qhat, khat, stats);

    sdp_finalize_kernel<<<1, 512, 0, stream>>>(stats, mr);

    sdp_attn_kernel<<<dim3((HH / 8) * (WW / 8), BATCH), 128, 0, stream>>>(
        x_low, qhat, khat, gq, bq, gk, bk, mr, out);
}
